// Linear4Bit_7078106104438
// MI455X (gfx1250) — matrix operand order repaired and checked by reading, not compiled, not run on hardware
//
#include <hip/hip_runtime.h>
#include <hip/hip_fp16.h>

#define TOKENS 4096
#define K_IN   4096
#define N_OUT  11008
#define GROUP  16
#define NBLOCKS (K_IN / GROUP)

#define M_BLK 128
#define N_BLK 128
#define K_TILE 32
#define LDSTRIDE 40   // padded bf16 stride: keeps 16B alignment, avoids bank conflicts

// wave tiling: 8 waves = 4(M) x 2(N); each wave computes 32(M) x 64(N)
#define WAVE_M_TILES 2
#define WAVE_N_TILES 4

typedef __attribute__((ext_vector_type(16))) __bf16        v16bf;
typedef __attribute__((ext_vector_type(8)))  float         v8f;
typedef __attribute__((ext_vector_type(4)))  unsigned int  u32x4;
typedef __attribute__((ext_vector_type(4)))  int           i32x4;
typedef __attribute__((ext_vector_type(4)))  float         f32x4;

union FragBF16 {
    u32x4 u[2];
    v16bf v;
    unsigned short e[16];
};

union Half8 {
    u32x4 u;
    unsigned short e[8];
};

// fp32 -> bf16 with round-to-nearest-even (pure bit math; no scalar __bf16 ops)
__device__ __forceinline__ unsigned short f2bf(float f) {
    unsigned int u = __float_as_uint(f);
    u += 0x7FFFu + ((u >> 16) & 1u);
    return (unsigned short)(u >> 16);
}

// ---------------------------------------------------------------------------
// Kernel 1: x fp32 -> bf16  (8 elements / thread)
// ---------------------------------------------------------------------------
__global__ __launch_bounds__(256) void convert_x_kernel(const float* __restrict__ x,
                                                        unsigned short* __restrict__ xb) {
    size_t t = (size_t)blockIdx.x * 256 + threadIdx.x;
    size_t base = t * 8;
    const f32x4* src = (const f32x4*)(x + base);
    f32x4 f0 = src[0];
    f32x4 f1 = src[1];
    Half8 o;
#pragma unroll
    for (int i = 0; i < 4; ++i) {
        o.e[i]     = f2bf(f0[i]);
        o.e[4 + i] = f2bf(f1[i]);
    }
    *(u32x4*)(xb + base) = o.u;
}

// ---------------------------------------------------------------------------
// Kernel 2: dequantize int4 groups -> bf16 weights [N_OUT, K_IN]
// one thread per (out, block) group of 16 elements
// ---------------------------------------------------------------------------
__global__ __launch_bounds__(256) void dequant_kernel(const int* __restrict__ wq,
                                                      const __half* __restrict__ wn,
                                                      unsigned short* __restrict__ wb) {
    size_t t = (size_t)blockIdx.x * 256 + threadIdx.x;   // over N_OUT * NBLOCKS
    const i32x4* q = (const i32x4*)(wq + t * 8);
    i32x4 q0 = q[0];
    i32x4 q1 = q[1];
    float norm = __half2float(wn[t]);
    float s = norm * (2.0f / 15.0f);

    int qq[8] = {q0[0], q0[1], q0[2], q0[3], q1[0], q1[1], q1[2], q1[3]};
    FragBF16 o;
#pragma unroll
    for (int i = 0; i < 8; ++i) {
        o.e[2 * i]     = f2bf((float)(qq[i] & 15) * s - norm);        // even (low nibble)
        o.e[2 * i + 1] = f2bf((float)((qq[i] >> 4) & 15) * s - norm); // odd (high nibble)
    }
    u32x4* dst = (u32x4*)(wb + t * 16);
    dst[0] = o.u[0];
    dst[1] = o.u[1];
}

// ---------------------------------------------------------------------------
// Kernel 3: WMMA bf16 GEMM  out[M,N] = Xb[M,K] * Wb[N,K]^T + bias
// 256 threads = 8 waves (wave32) as 4(M) x 2(N); each wave -> 32x64 output
// via 2x4 v_wmma_f32_16x16x32_bf16 accumulators (A-fragment reused over 4 Ns).
// Block tile 128(M) x 128(N), K step 32.
// ---------------------------------------------------------------------------
__global__ __launch_bounds__(256) void gemm_wmma_kernel(const unsigned short* __restrict__ Xb,
                                                        const unsigned short* __restrict__ Wb,
                                                        const float* __restrict__ bias,
                                                        float* __restrict__ out) {
    __shared__ __align__(16) unsigned short Xs[M_BLK * LDSTRIDE];
    __shared__ __align__(16) unsigned short Ws[N_BLK * LDSTRIDE];

    const int tid  = threadIdx.x;
    const int wave = tid >> 5;
    const int lane = tid & 31;
    const int wm   = wave >> 1;   // 0..3  (M direction, 32 rows each)
    const int wn   = wave & 1;    // 0..1  (N direction, 64 cols each)

    const int mBase = blockIdx.y * M_BLK;
    const int nBase = blockIdx.x * N_BLK;

    // cooperative tile loaders: 4 threads per row, 8 bf16 (16B) each; 2 passes/matrix
    const int lrow = tid >> 2;         // 0..63
    const int lcol = (tid & 3) << 3;   // 0,8,16,24

    // CDNA5 16-bit fragment lane mapping (ISA 7.12.2)
    const int laneM = lane & 15;
    const int kSelA = (lane >> 4) << 3;   // A: lanes 16-31 hold K+8 within each 16-chunk
    const int kSelB = (lane >> 4) << 3;   // B: lanes 16-31 hold K+8 within each 16-chunk

    v8f acc[WAVE_M_TILES][WAVE_N_TILES] = {};

    for (int k0 = 0; k0 < K_IN; k0 += K_TILE) {
        // ---- global -> LDS staging (16B vector moves), 128 rows per matrix ----
        const size_t xoff0 = (size_t)(mBase + lrow) * K_IN + k0 + lcol;
        const size_t xoff1 = (size_t)(mBase + 64 + lrow) * K_IN + k0 + lcol;
        *(u32x4*)(&Xs[lrow * LDSTRIDE + lcol])        = *(const u32x4*)(Xb + xoff0);
        *(u32x4*)(&Xs[(64 + lrow) * LDSTRIDE + lcol]) = *(const u32x4*)(Xb + xoff1);

        const size_t woff0 = (size_t)(nBase + lrow) * K_IN + k0 + lcol;
        const size_t woff1 = (size_t)(nBase + 64 + lrow) * K_IN + k0 + lcol;
        *(u32x4*)(&Ws[lrow * LDSTRIDE + lcol])        = *(const u32x4*)(Wb + woff0);
        *(u32x4*)(&Ws[(64 + lrow) * LDSTRIDE + lcol]) = *(const u32x4*)(Wb + woff1);

        // prefetch next K tile into the cache hierarchy (global_prefetch_b8)
        if (k0 + K_TILE < K_IN) {
            __builtin_prefetch((const void*)(Xb + xoff0 + K_TILE), 0, 1);
            __builtin_prefetch((const void*)(Xb + xoff1 + K_TILE), 0, 1);
            __builtin_prefetch((const void*)(Wb + woff0 + K_TILE), 0, 1);
            __builtin_prefetch((const void*)(Wb + woff1 + K_TILE), 0, 1);
        }

        __syncthreads();

        // ---- build WMMA fragments from LDS ----
        FragBF16 a[WAVE_M_TILES], b[WAVE_N_TILES];
#pragma unroll
        for (int t = 0; t < WAVE_M_TILES; ++t) {
            const int m = wm * 32 + t * 16 + laneM;
            // A (16x32 bf16): chunk0 = K[kSelA .. kSelA+7], chunk1 = K[16+kSelA .. +7]
            a[t].u[0] = *(const u32x4*)(&Xs[m * LDSTRIDE + kSelA]);
            a[t].u[1] = *(const u32x4*)(&Xs[m * LDSTRIDE + 16 + kSelA]);
        }
#pragma unroll
        for (int t = 0; t < WAVE_N_TILES; ++t) {
            const int n = wn * 64 + t * 16 + laneM;
            // B (32x16 bf16): chunk0 = K[kSelB .. kSelB+7], chunk1 = K[16+kSelB .. +7]
            b[t].u[0] = *(const u32x4*)(&Ws[n * LDSTRIDE + kSelB]);
            b[t].u[1] = *(const u32x4*)(&Ws[n * LDSTRIDE + 16 + kSelB]);
        }

        // ---- 8 WMMAs: D = A*B + C  (A reused across the 4 N tiles) ----
#pragma unroll
        for (int tm = 0; tm < WAVE_M_TILES; ++tm)
#pragma unroll
            for (int tn = 0; tn < WAVE_N_TILES; ++tn)
                acc[tm][tn] = __builtin_amdgcn_wmma_f32_16x16x32_bf16(
                    /*neg_a=*/false, a[tm].v,
                    /*neg_b=*/false, b[tn].v,
                    /*c_mod=*/(short)0, acc[tm][tn],
                    /*reuse_a=*/false, /*reuse_b=*/false);

        __syncthreads();
    }

    // ---- epilogue: C/D layout -> memory, add bias ----
    const int rowHalf = (lane >> 4) << 3;   // VGPR r holds M=r (lanes 0-15) / M=r+8 (16-31)
#pragma unroll
    for (int tm = 0; tm < WAVE_M_TILES; ++tm) {
#pragma unroll
        for (int tn = 0; tn < WAVE_N_TILES; ++tn) {
            const int row0 = mBase + wm * 32 + tm * 16 + rowHalf;
            const int col  = nBase + wn * 64 + tn * 16 + laneM;
            const float bv = bias[col];
#pragma unroll
            for (int r = 0; r < 8; ++r) {
                out[(size_t)(row0 + r) * N_OUT + col] = acc[tm][tn][r] + bv;
            }
        }
    }
}

// ---------------------------------------------------------------------------
extern "C" void kernel_launch(void* const* d_in, const int* in_sizes, int n_in,
                              void* d_out, int out_size, void* d_ws, size_t ws_size,
                              hipStream_t stream) {
    (void)in_sizes; (void)n_in; (void)out_size; (void)ws_size;

    const float*  x    = (const float*)d_in[0];
    const int*    wq   = (const int*)d_in[1];
    const __half* wn   = (const __half*)d_in[2];
    const float*  bias = (const float*)d_in[3];
    float*        out  = (float*)d_out;

    unsigned short* Xb = (unsigned short*)d_ws;                   // [TOKENS, K_IN] bf16
    unsigned short* Wb = Xb + (size_t)TOKENS * K_IN;              // [N_OUT,  K_IN] bf16

    // 1) x -> bf16
    {
        const int total = TOKENS * K_IN;                           // 16,777,216
        const int blocks = total / (256 * 8);                      // 8192
        convert_x_kernel<<<blocks, 256, 0, stream>>>(x, Xb);
    }
    // 2) dequantize weights -> bf16
    {
        const int groups = N_OUT * NBLOCKS;                        // 2,818,048
        const int blocks = groups / 256;                           // 11008
        dequant_kernel<<<blocks, 256, 0, stream>>>(wq, wn, Wb);
    }
    // 3) WMMA GEMM + bias
    {
        dim3 grid(N_OUT / N_BLK, TOKENS / M_BLK);                  // (86, 32)
        gemm_wmma_kernel<<<grid, 256, 0, stream>>>(Xb, Wb, bias, out);
    }
}
